// MMoE_11158325035641
// MI455X (gfx1250) — compile-verified, matrix operand order checked by reading, not run on hardware
//
#include <hip/hip_runtime.h>
#include <hip/hip_bf16.h>

// Problem sizes (fixed by the reference)
#define BB 16384
#define DD 1024
#define UU 512
#define EE 8
#define TT 4

// Tiling
#define BM 64               // rows per block
#define BN 64               // U columns per block
#define KC 32               // K per WMMA step (bf16 16x16x32)
#define NSTEP (DD / KC)     // 32 k-steps per expert
#define NT (EE * NSTEP)     // 256 total steps
#define XPITCH (DD + 8)     // bf16 elems per x row in LDS (pad 16B, keeps 16B align)
#define WPITCH (KC + 8)     // bf16 elems per w row in LDS (80B rows -> bank spread)

typedef __attribute__((ext_vector_type(16))) __bf16 v16bf;
typedef __attribute__((ext_vector_type(8)))  __bf16 v8bf;
typedef __attribute__((ext_vector_type(4)))  __bf16 v4bf;
typedef __attribute__((ext_vector_type(2)))  __bf16 v2bf;
typedef __attribute__((ext_vector_type(8)))  float  v8f;
typedef __attribute__((ext_vector_type(4)))  float  v4f;
typedef __attribute__((ext_vector_type(2)))  float  v2f;

#if __has_builtin(__builtin_amdgcn_global_load_async_to_lds_b128) && \
    __has_builtin(__builtin_amdgcn_s_wait_asynccnt)
#define HAVE_ASYNC 1
#else
#define HAVE_ASYNC 0
#endif

static __device__ __forceinline__ v16bf frag16(const __bf16* p) {
    // 16 bf16 per lane = two contiguous 16B LDS reads (K k..k+7 and K k+16..k+23)
    v8bf lo = *(const v8bf*)(p);
    v8bf hi = *(const v8bf*)(p + 16);
    return __builtin_shufflevector(lo, hi, 0,1,2,3,4,5,6,7,8,9,10,11,12,13,14,15);
}

static __device__ __forceinline__ v8f bfwmma(v16bf a, v16bf b, v8f c) {
    // D = A(16x32 bf16) * B(32x16 bf16) + C(16x16 f32)
    return __builtin_amdgcn_wmma_f32_16x16x32_bf16(
        /*neg_a=*/false, a, /*neg_b=*/false, b,
        /*c_mod=*/(short)0, c, /*reuse_a=*/false, /*reuse_b=*/false);
}

// Split a (k2, k2+1) f32 pair into bf16 hi/lo and store packed into the
// transposed [u][k] LDS planes. convertvector -> v_cvt_pk_bf16_f32.
static __device__ __forceinline__ void splitStore(float w0i, float w1i,
                                                  __bf16* phi, __bf16* plo) {
    const v2f hv = {w0i, w1i};
    const v2bf h = __builtin_convertvector(hv, v2bf);
    const v2f lv = hv - __builtin_convertvector(h, v2f);
    const v2bf l = __builtin_convertvector(lv, v2bf);
    *(v2bf*)phi = h;
    *(v2bf*)plo = l;
}

#if HAVE_ASYNC
// Builtin signature (from hipcc diagnostic): v4i addrspace(1)*, v4i addrspace(3)*, imm, imm
typedef int v4i_vs __attribute__((vector_size(16)));
typedef __attribute__((address_space(1))) v4i_vs GV4;
typedef __attribute__((address_space(3))) v4i_vs LV4;

// Async DMA 16B global -> LDS (ASYNCcnt tracked, no VGPR round-trip)
static __device__ __forceinline__ void asyncCopy16(const void* g, void* l) {
    __builtin_amdgcn_global_load_async_to_lds_b128((GV4*)g, (LV4*)l, 0, 0);
}

// Issue async loads of the raw f32 weight tile for step t into fbuf[t&1].
// Lane-consecutive in U -> coalesced global addresses.
static __device__ __forceinline__ void issueAsync(const float* __restrict__ wek,
                                                  int u0, int t, int tid,
                                                  float (*fbuf)[KC][BN]) {
    const int e   = t >> 5;
    const int kc  = t & 31;
    const int buf = t & 1;
    const int k2  = (tid >> 4) << 1;    // even k in [0,30]
    const int ug  = (tid & 15) << 2;    // u group of 4 in [0,60]
    const float* gp = wek + (size_t)e * DD * UU + (size_t)(kc * KC + k2) * UU + (u0 + ug);
    asyncCopy16(gp,      &fbuf[buf][k2][ug]);
    asyncCopy16(gp + UU, &fbuf[buf][k2 + 1][ug]);
}

// Convert the f32 tile for step t (already landed in fbuf[t&1] by this
// thread's own async loads) into bf16 hi/lo planes, transposed [u][k].
static __device__ __forceinline__ void convertTile(int t, int tid,
                                                   float (*fbuf)[KC][BN],
                                                   __bf16 (*whi)[BN][WPITCH],
                                                   __bf16 (*wlo)[BN][WPITCH]) {
    const int buf = t & 1;
    const int k2  = (tid >> 4) << 1;
    const int ug  = (tid & 15) << 2;
    const float4 w0 = *(const float4*)&fbuf[buf][k2][ug];
    const float4 w1 = *(const float4*)&fbuf[buf][k2 + 1][ug];
    const float* a0 = (const float*)&w0;
    const float* a1 = (const float*)&w1;
#pragma unroll
    for (int i = 0; i < 4; ++i)
        splitStore(a0[i], a1[i], &whi[buf][ug + i][k2], &wlo[buf][ug + i][k2]);
}
#else
// Synchronous fallback: global f32 tile -> bf16 hi/lo planes, transposed.
static __device__ __forceinline__ void stageW(const float* __restrict__ wek,
                                              int u0, int t, int tid,
                                              __bf16 (*whi)[BN][WPITCH],
                                              __bf16 (*wlo)[BN][WPITCH]) {
    const int e   = t >> 5;
    const int kc  = t & 31;
    const int buf = t & 1;
    const int k2  = (tid >> 4) << 1;
    const int ug  = (tid & 15) << 2;
    const float* gp = wek + (size_t)e * DD * UU + (size_t)(kc * KC + k2) * UU + (u0 + ug);
    const float4 w0 = *(const float4*)(gp);
    const float4 w1 = *(const float4*)(gp + UU);
    const float* a0 = (const float*)&w0;
    const float* a1 = (const float*)&w1;
#pragma unroll
    for (int i = 0; i < 4; ++i)
        splitStore(a0[i], a1[i], &whi[buf][ug + i][k2], &wlo[buf][ug + i][k2]);
}
#endif

__global__ __launch_bounds__(256)
void mmoe_fused_kernel(const float* __restrict__ x,      // [B,D]
                       const float* __restrict__ wek,    // [E,D,U]
                       const float* __restrict__ ebias,  // [E,U]
                       const float* __restrict__ gk,     // [T,D,E]
                       const float* __restrict__ gbias,  // [T,E]
                       float* __restrict__ out) {        // [T,B,U]
    __shared__ __bf16 xhi[BM][XPITCH];          // 129 KB
    __shared__ __bf16 xlo[BM][XPITCH];          // 129 KB
    __shared__ __bf16 whi[2][BN][WPITCH];       // 10 KB (double buffered)
    __shared__ __bf16 wlo[2][BN][WPITCH];       // 10 KB
    __shared__ float  gsm[BM][TT][EE];          // 8 KB softmax gates
#if HAVE_ASYNC
    __shared__ float  fbuf[2][KC][BN];          // 16 KB raw f32 async landing pads
#endif

    const int tid  = threadIdx.x;
    const int row0 = blockIdx.x * BM;
    const int u0   = blockIdx.y * BN;

#if HAVE_ASYNC
    // Prime the async pipeline: tiles 0 and 1 in flight immediately.
    issueAsync(wek, u0, 0, tid, fbuf);
    issueAsync(wek, u0, 1, tid, fbuf);
#endif

    // ---- Prologue A: stage x stripe (64 x 1024) as bf16 hi/lo planes ----
#pragma unroll 4
    for (int j = 0; j < (BM * DD / 4) / 256; ++j) {   // 64 float4s per thread
        const int f = tid + j * 256;
        const int r = f >> 8;             // DD/4 = 256 float4 per row
        const int c = (f & 255) << 2;     // d offset
        const float4 v = *(const float4*)(x + (size_t)(row0 + r) * DD + c);
        const float* va = (const float*)&v;
        const v4f vf = {va[0], va[1], va[2], va[3]};
        const v4bf h = __builtin_convertvector(vf, v4bf);
        const v4f lf = vf - __builtin_convertvector(h, v4f);
        const v4bf l = __builtin_convertvector(lf, v4bf);
        *(v4bf*)&xhi[r][c] = h;
        *(v4bf*)&xlo[r][c] = l;
    }

    // ---- Prologue B: gate logits + softmax (one thread per (row, task)) ----
    {
        const int gr = tid >> 2;                    // row 0..63
        const int gt = tid & 3;                     // task 0..3
        const float* xr  = x + (size_t)(row0 + gr) * DD;
        const float* gkb = gk + (size_t)gt * DD * EE;
        float acc[EE];
#pragma unroll
        for (int e = 0; e < EE; ++e) acc[e] = gbias[gt * EE + e];
        for (int d = 0; d < DD; ++d) {
            const float xv = xr[d];
            const float4 g0 = *(const float4*)(gkb + (size_t)d * EE);
            const float4 g1 = *(const float4*)(gkb + (size_t)d * EE + 4);
            const float* g0a = (const float*)&g0;
            const float* g1a = (const float*)&g1;
#pragma unroll
            for (int e = 0; e < 4; ++e) {
                acc[e]     += xv * g0a[e];
                acc[e + 4] += xv * g1a[e];
            }
        }
        float m = acc[0];
#pragma unroll
        for (int e = 1; e < EE; ++e) m = fmaxf(m, acc[e]);
        float s = 0.f;
#pragma unroll
        for (int e = 0; e < EE; ++e) { acc[e] = __expf(acc[e] - m); s += acc[e]; }
        const float inv = 1.f / s;
#pragma unroll
        for (int e = 0; e < EE; ++e) gsm[gr][gt][e] = acc[e] * inv;
    }

#if HAVE_ASYNC
    __builtin_amdgcn_s_wait_asynccnt(2);        // tile 0 landed (tile 1 in flight)
    convertTile(0, tid, fbuf, whi, wlo);
#else
    stageW(wek, u0, 0, tid, whi, wlo);
#endif
    __syncthreads();

    // ---- Lane/wave geometry ----
    const int lane = tid & 31;
    const int wid  = tid >> 5;
    const int wm   = wid >> 1;          // 0..3 -> row offset 16*wm
    const int wn   = wid & 1;           // 0..1 -> col offset 32*wn
    const int half = lane >> 4;         // K-half select per ISA layout
    const int l16  = lane & 15;
    const int kb   = half << 3;         // 0 or 8
    const int rA   = wm * 16 + l16;     // A-matrix row (local)
    const int cB0  = wn * 32 + l16;     // B-matrix column, tile 0
    const int cB1  = cB0 + 16;          // tile 1

    const v8f vzero = {0.f, 0.f, 0.f, 0.f, 0.f, 0.f, 0.f, 0.f};
    v8f hAcc0 = vzero, hAcc1 = vzero;
    v8f tAcc[TT][2];
#pragma unroll
    for (int t = 0; t < TT; ++t) { tAcc[t][0] = vzero; tAcc[t][1] = vzero; }

    // ---- Main loop: experts x k-chunks with pipelined weight staging ----
    for (int step = 0; step < NT; ++step) {
        const int buf = step & 1;
        const int k0  = (step & 31) * KC;

        const v16bf ah  = frag16(&xhi[rA][k0 + kb]);
        const v16bf al  = frag16(&xlo[rA][k0 + kb]);
        const v16bf bh0 = frag16(&whi[buf][cB0][kb]);
        const v16bf bl0 = frag16(&wlo[buf][cB0][kb]);
        const v16bf bh1 = frag16(&whi[buf][cB1][kb]);
        const v16bf bl1 = frag16(&wlo[buf][cB1][kb]);

        // bf16x3 split-precision accumulation (hi*hi + hi*lo + lo*hi)
        hAcc0 = bfwmma(ah, bh0, hAcc0);
        hAcc0 = bfwmma(ah, bl0, hAcc0);
        hAcc0 = bfwmma(al, bh0, hAcc0);
        hAcc1 = bfwmma(ah, bh1, hAcc1);
        hAcc1 = bfwmma(ah, bl1, hAcc1);
        hAcc1 = bfwmma(al, bh1, hAcc1);

#if HAVE_ASYNC
        if (step + 2 < NT) {
            issueAsync(wek, u0, step + 2, tid, fbuf);   // fbuf[step&1] is free now
            __builtin_amdgcn_s_wait_asynccnt(2);        // tile step+1 landed
            convertTile(step + 1, tid, fbuf, whi, wlo);
        } else if (step + 1 < NT) {
            __builtin_amdgcn_s_wait_asynccnt(0);
            convertTile(step + 1, tid, fbuf, whi, wlo);
        }
#else
        if (step + 1 < NT) stageW(wek, u0, step + 1, tid, whi, wlo);
#endif

        if ((step & 31) == 31) {
            // Finished expert e: bias + ReLU + gated fold into task accumulators
            const int e = step >> 5;
            const float eb0 = ebias[e * UU + u0 + cB0];
            const float eb1 = ebias[e * UU + u0 + cB1];
#pragma unroll
            for (int i = 0; i < 8; ++i) {
                const int rl = wm * 16 + (half << 3) + i;  // C row = i + 8*half
                const float h0 = fmaxf(hAcc0[i] + eb0, 0.f);
                const float h1 = fmaxf(hAcc1[i] + eb1, 0.f);
#pragma unroll
                for (int t = 0; t < TT; ++t) {
                    const float gv = gsm[rl][t][e];        // LDS broadcast read
                    tAcc[t][0][i] += gv * h0;
                    tAcc[t][1][i] += gv * h1;
                }
            }
            hAcc0 = vzero;
            hAcc1 = vzero;
        }
        __syncthreads();
    }

    // ---- Epilogue: coalesced stores of out[t, b, u] ----
#pragma unroll
    for (int t = 0; t < TT; ++t) {
        float* ob = out + (size_t)t * BB * UU;
#pragma unroll
        for (int i = 0; i < 8; ++i) {
            const int r = row0 + wm * 16 + (half << 3) + i;
            ob[(size_t)r * UU + u0 + cB0] = tAcc[t][0][i];
            ob[(size_t)r * UU + u0 + cB1] = tAcc[t][1][i];
        }
    }
}

extern "C" void kernel_launch(void* const* d_in, const int* in_sizes, int n_in,
                              void* d_out, int out_size, void* d_ws, size_t ws_size,
                              hipStream_t stream) {
    const float* x     = (const float*)d_in[0];  // [B,D]
    const float* wek   = (const float*)d_in[1];  // [E,D,U]
    const float* ebias = (const float*)d_in[2];  // [E,U]
    const float* gk    = (const float*)d_in[3];  // [T,D,E]
    const float* gbias = (const float*)d_in[4];  // [T,E]
    float* out = (float*)d_out;                  // [T,B,U]

    dim3 grid(BB / BM, UU / BN);   // 256 x 8
    dim3 block(256);
    mmoe_fused_kernel<<<grid, block, 0, stream>>>(x, wek, ebias, gk, gbias, out);
}
